// EquivariantPerturbationTransform_41360535060903
// MI455X (gfx1250) — matrix operand order repaired and checked by reading, not compiled, not run on hardware
//
#include <hip/hip_runtime.h>
#include <cstddef>
#include <cstdint>

// ---------------------------------------------------------------------------
// CDNA5 (gfx1250) WMMA types & helpers
// ---------------------------------------------------------------------------
typedef __bf16 v16bf __attribute__((ext_vector_type(16)));
typedef float  v8f   __attribute__((ext_vector_type(8)));

struct __attribute__((aligned(16))) U128 { unsigned int x, y, z, w; };
union AB16 { v16bf v; U128 q[2]; };

__device__ __forceinline__ v8f wmma_bf16(v16bf a, v16bf b, v8f c) {
  // D = A(16x32) * B(32x16) + C(16x16), f32 accumulate
  return __builtin_amdgcn_wmma_f32_16x16x32_bf16(false, a, false, b,
                                                 (short)0, c, false, false);
}

// A fragment: lane holds row (m), K chunks [k0+h*8 .. +8) and [k0+h*8+16 .. +8)
__device__ __forceinline__ v16bf load_a(const __bf16* A, int lda, int row,
                                        int k0, int h) {
  AB16 f;
  const __bf16* p = A + (size_t)row * lda + k0 + h * 8;
  f.q[0] = *(const U128*)(p);
  f.q[1] = *(const U128*)(p + 16);
  return f.v;
}
// B fragment: lane holds column (n), K chunks [k0+h*8 .. +8) and [k0+h*8+16 .. +8)
__device__ __forceinline__ v16bf load_b(const __bf16* B, int ldb, int col,
                                        int k0, int h) {
  AB16 f;
  const __bf16* p = B + (size_t)col * ldb + k0 + h * 8;
  f.q[0] = *(const U128*)(p);
  f.q[1] = *(const U128*)(p + 16);
  return f.v;
}

// ---------------------------------------------------------------------------
// Async global->LDS copy (CDNA5 GLOBAL_LOAD_ASYNC_TO_LDS_B128, ASYNCcnt).
// LDS operand is the low 32 bits of the flat address (ISA 10.2: LDS aperture
// address maps to addr[31:0]).
// ---------------------------------------------------------------------------
__device__ __forceinline__ void async_copy16(const void* gsrc, void* ldst) {
  unsigned loff = (unsigned)(uintptr_t)ldst;
  asm volatile("global_load_async_to_lds_b128 %0, %1, off"
               :
               : "v"(loff), "v"((unsigned long long)(uintptr_t)gsrc)
               : "memory");
}
__device__ __forceinline__ void wait_async_le6() {
#if __has_builtin(__builtin_amdgcn_s_wait_asynccnt)
  __builtin_amdgcn_s_wait_asynccnt(6);
#else
  asm volatile("s_wait_asynccnt 0x6" ::: "memory");
#endif
}
__device__ __forceinline__ void wait_async_le0() {
#if __has_builtin(__builtin_amdgcn_s_wait_asynccnt)
  __builtin_amdgcn_s_wait_asynccnt(0);
#else
  asm volatile("s_wait_asynccnt 0x0" ::: "memory");
#endif
}

// ---------------------------------------------------------------------------
// Small utility kernels
// ---------------------------------------------------------------------------
__global__ void cvt_kernel(const float* __restrict__ s, __bf16* __restrict__ d,
                           int n) {
  for (int i = blockIdx.x * blockDim.x + threadIdx.x; i < n;
       i += gridDim.x * blockDim.x)
    d[i] = (__bf16)s[i];
}

__global__ void count_kernel(const int* __restrict__ bas, int* __restrict__ cnt) {
  int b = threadIdx.x;
  if (b < 16) {
    int c = 0;
    for (int p = 0; p < 128; ++p) c += (bas[p] == b) ? 1 : 0;
    cnt[b] = c;
  }
}

__global__ void gather_kernel(const __bf16* __restrict__ Hb,
                              const int* __restrict__ idx,
                              __bf16* __restrict__ Hp) {
  int p = blockIdx.x, c = threadIdx.x;   // <<<128, 256>>>
  Hp[(size_t)p * 256 + c] = Hb[(size_t)idx[p] * 256 + c];
}

// Vt[h][e][p] = V[p][h*32+e]   (bf16, 8*32*128)
__global__ void vtrans_kernel(const __bf16* __restrict__ V,
                              __bf16* __restrict__ Vt) {
  int i = blockIdx.x * blockDim.x + threadIdx.x;
  if (i < 8 * 32 * 128) {
    int hh = i >> 12, e = (i >> 7) & 31, p = i & 127;
    Vt[i] = V[(size_t)p * 256 + hh * 32 + e];
  }
}

// ---------------------------------------------------------------------------
// Epilogue helper. EPI: 0 = none, 1 = exact GELU.
// ---------------------------------------------------------------------------
template <int EPI>
__device__ __forceinline__ void store_tile(__bf16* C, int ldc, int M, int mb,
                                           int nb, int l, int h, v8f acc,
                                           const float* bias) {
  int col = nb + l;
  float bv = bias[col];
#pragma unroll
  for (int r = 0; r < 8; ++r) {
    int row = mb + r + 8 * h;
    if (row < M) {
      float v = acc[r] + bv;
      if (EPI == 1) v = 0.5f * v * (1.0f + erff(v * 0.70710678118654752f));
      C[(size_t)row * ldc + col] = (__bf16)v;
    }
  }
}

// ---------------------------------------------------------------------------
// Pipelined GEMM:  C[M,Nc] = epi(A[M,K] @ W[Nc,K]^T + bias), bf16 in/out.
// Block tile 64x128, K-stage 64, double-buffered LDS filled with
// GLOBAL_LOAD_ASYNC_TO_LDS_B128; 8 waves, each a 32x32 tile (2x2 WMMA).
// grid.x = ceil(M/64), grid.y = Nc/128. K must be a multiple of 64.
// ---------------------------------------------------------------------------
__device__ __forceinline__ void issue_stage(const __bf16* A, int lda,
                                            int m_base, int M,
                                            const __bf16* W, int ldb,
                                            int n_base, int k0, __bf16* sA,
                                            __bf16* sB, int t) {
  // A tile: 64 rows x 64 bf16 = 512 chunks of 16B -> 2 per thread
#pragma unroll
  for (int i = 0; i < 2; ++i) {
    int c = t + i * 256;
    int row = c >> 3;
    int cc = (c & 7) * 8;
    int rg = m_base + row;
    if (rg > M - 1) rg = M - 1;  // clamp keeps addresses valid, EXEC uniform
    async_copy16(A + (size_t)rg * lda + k0 + cc, sA + row * 64 + cc);
  }
  // B tile: 128 rows x 64 bf16 = 1024 chunks of 16B -> 4 per thread
#pragma unroll
  for (int i = 0; i < 4; ++i) {
    int c = t + i * 256;
    int row = c >> 3;
    int cc = (c & 7) * 8;
    async_copy16(W + (size_t)(n_base + row) * ldb + k0 + cc,
                 sB + row * 64 + cc);
  }
}

template <int EPI>
__global__ __launch_bounds__(256) void gemm_async_kernel(
    const __bf16* __restrict__ A, int lda, const __bf16* __restrict__ W,
    int ldb, const float* __restrict__ bias, __bf16* __restrict__ C, int ldc,
    int M, int K) {
  __shared__ __bf16 sA[2][64 * 64];    // 2 x 8 KB
  __shared__ __bf16 sB[2][128 * 64];   // 2 x 16 KB
  const int t = threadIdx.x, wave = t >> 5, lane = t & 31;
  const int l = lane & 15, h = lane >> 4;
  const int wm = wave >> 2, wn = wave & 3;
  const int m_blk = blockIdx.x * 64, n_blk = blockIdx.y * 128;
  const int nstages = K / 64;

  issue_stage(A, lda, m_blk, M, W, ldb, n_blk, 0, sA[0], sB[0], t);

  v8f acc00 = {}, acc01 = {}, acc10 = {}, acc11 = {};
  int buf = 0;
  for (int s = 0; s < nstages; ++s) {
    if (s + 1 < nstages) {
      issue_stage(A, lda, m_blk, M, W, ldb, n_blk, (s + 1) * 64, sA[buf ^ 1],
                  sB[buf ^ 1], t);
      wait_async_le6();   // stage s's 6 copies done; s+1's 6 may be in flight
    } else {
      wait_async_le0();
    }
    __syncthreads();      // all waves' copies for stage s visible in LDS
    const __bf16* a_base = sA[buf] + (wm * 32) * 64;
    const __bf16* b_base = sB[buf] + (wn * 32) * 64;
#pragma unroll
    for (int kk = 0; kk < 64; kk += 32) {
      v16bf a0 = load_a(a_base, 64, l, kk, h);
      v16bf a1 = load_a(a_base, 64, 16 + l, kk, h);
      v16bf b0 = load_b(b_base, 64, l, kk, h);
      v16bf b1 = load_b(b_base, 64, 16 + l, kk, h);
      acc00 = wmma_bf16(a0, b0, acc00);
      acc01 = wmma_bf16(a0, b1, acc01);
      acc10 = wmma_bf16(a1, b0, acc10);
      acc11 = wmma_bf16(a1, b1, acc11);
    }
    __syncthreads();      // done reading buf before next issue overwrites it
    buf ^= 1;
  }
  const int m0 = m_blk + wm * 32, n0 = n_blk + wn * 32;
  store_tile<EPI>(C, ldc, M, m0,      n0,      l, h, acc00, bias);
  store_tile<EPI>(C, ldc, M, m0,      n0 + 16, l, h, acc01, bias);
  store_tile<EPI>(C, ldc, M, m0 + 16, n0,      l, h, acc10, bias);
  store_tile<EPI>(C, ldc, M, m0 + 16, n0 + 16, l, h, acc11, bias);
}

// ---------------------------------------------------------------------------
// scores[h][n][p] = scale * Q_h[n,:] . K_h[p,:]   (K = 32, one WMMA per wave)
// grid = (N/16, 8 heads), block = 256 (8 waves cover all 128 p)
// ---------------------------------------------------------------------------
__global__ __launch_bounds__(256) void scores_kernel(
    const __bf16* __restrict__ Qb, const __bf16* __restrict__ Kb,
    float* __restrict__ S, int M) {
  const int t = threadIdx.x, wave = t >> 5, lane = t & 31;
  const int l = lane & 15, h = lane >> 4;
  const int m0 = blockIdx.x * 16;
  const int head = blockIdx.y;
  const int p0 = wave * 16;
  v16bf a  = load_a(Qb + head * 32, 256, m0 + l, 0, h);
  v16bf bb = load_b(Kb + head * 32, 256, p0 + l, 0, h);
  v8f acc = {};
  acc = wmma_bf16(a, bb, acc);
  const float scale = 0.17677669529663687f;  // 1/sqrt(32)
#pragma unroll
  for (int r = 0; r < 8; ++r) {
    int row = m0 + r + 8 * h;
    S[((size_t)head * M + row) * 128 + p0 + l] = acc[r] * scale;
  }
}

// ---------------------------------------------------------------------------
// Per-batch fused masked softmax (over p) + ctx = probs @ V_h.
// Block handles 16 query rows, all 8 heads (wave == head). grid = N/16.
// ---------------------------------------------------------------------------
__global__ __launch_bounds__(256) void attn_ctx_kernel(
    const float* __restrict__ S, const __bf16* __restrict__ Vt,
    const int* __restrict__ bas, const int* __restrict__ cnt, int b,
    __bf16* __restrict__ ctx, int M) {
  __shared__ __bf16 sP[8][16][128];   // probs, bf16, 32 KB
  __shared__ int sMask[128];
  const int t = threadIdx.x;
  if (t < 128) sMask[t] = (bas[t] == b) ? 1 : 0;
  __syncthreads();
  const int m0 = blockIdx.x * 16;
  if (t < 128) {  // one thread per (head, row)
    int hh = t >> 4, r = t & 15;
    const float* srow = S + ((size_t)hh * M + m0 + r) * 128;
    float mx = -1e30f;
    for (int p = 0; p < 128; ++p)
      if (sMask[p]) mx = fmaxf(mx, srow[p]);
    float sum = 0.f;
    for (int p = 0; p < 128; ++p) {
      float e = sMask[p] ? __expf(srow[p] - mx) : 0.f;
      sum += e;
      sP[hh][r][p] = (__bf16)e;
    }
    float inv = (cnt[b] > 0) ? (1.f / sum) : 0.f;  // empty batch -> zeros
    for (int p = 0; p < 128; ++p)
      sP[hh][r][p] = (__bf16)((float)sP[hh][r][p] * inv);
  }
  __syncthreads();
  const int wave = t >> 5, lane = t & 31, l = lane & 15, h = lane >> 4;
  v8f acc0 = {}, acc1 = {};
  const __bf16* Vh = Vt + (size_t)wave * 32 * 128;  // [32 e][128 p] row-major
  for (int k0 = 0; k0 < 128; k0 += 32) {
    AB16 fa;
    const __bf16* pa = &sP[wave][l][k0 + h * 8];
    fa.q[0] = *(const U128*)(pa);
    fa.q[1] = *(const U128*)(pa + 16);
    v16bf b0 = load_b(Vh, 128, l, k0, h);
    v16bf b1 = load_b(Vh, 128, 16 + l, k0, h);
    acc0 = wmma_bf16(fa.v, b0, acc0);
    acc1 = wmma_bf16(fa.v, b1, acc1);
  }
#pragma unroll
  for (int r = 0; r < 8; ++r) {
    int row = m0 + r + 8 * h;
    ctx[(size_t)row * 256 + wave * 32 + l]      = (__bf16)acc0[r];
    ctx[(size_t)row * 256 + wave * 32 + 16 + l] = (__bf16)acc1[r];
  }
}

// ---------------------------------------------------------------------------
// Fused GEMM (Nc=256 full row) + residual + LayerNorm.
//   y = LN( resid + ha * (A @ W^T + bias) ) * g + beta
// ha = (cnt ? cnt[b]>0 : 1). Block = 16 rows x 256 cols; 8 waves of 16x32.
// Writes f32 (fout) and optionally bf16 (bout). grid = M/16.
// ---------------------------------------------------------------------------
__global__ __launch_bounds__(256) void gemm_ln_kernel(
    const __bf16* __restrict__ A, int lda, int K, const __bf16* __restrict__ W,
    const float* __restrict__ bias, const float* __restrict__ resid,
    const float* __restrict__ g, const float* __restrict__ beta,
    const int* __restrict__ cnt, int b, float* __restrict__ fout,
    __bf16* __restrict__ bout, int M) {
  __shared__ float xs[16][256];  // 16 KB
  __shared__ float smu[16], srs[16];
  const int t = threadIdx.x, wave = t >> 5, lane = t & 31;
  const int l = lane & 15, h = lane >> 4;
  const int m0 = blockIdx.x * 16;
  const int n0 = wave * 32;
  const int rA = m0 + l;  // M is a multiple of 16
  v8f acc0 = {}, acc1 = {};
  for (int k0 = 0; k0 < K; k0 += 32) {
    v16bf a  = load_a(A, lda, rA, k0, h);
    v16bf b0 = load_b(W, K, n0 + l, k0, h);
    v16bf b1 = load_b(W, K, n0 + 16 + l, k0, h);
    acc0 = wmma_bf16(a, b0, acc0);
    acc1 = wmma_bf16(a, b1, acc1);
  }
  const float ha = (cnt == nullptr) ? 1.f : (cnt[b] > 0 ? 1.f : 0.f);
#pragma unroll
  for (int r = 0; r < 8; ++r) {
    int rr = r + 8 * h, row = m0 + rr;
    int c0 = n0 + l, c1 = n0 + 16 + l;
    xs[rr][c0] = resid[(size_t)row * 256 + c0] + ha * (acc0[r] + bias[c0]);
    xs[rr][c1] = resid[(size_t)row * 256 + c1] + ha * (acc1[r] + bias[c1]);
  }
  __syncthreads();
  if (t < 16) {
    float mu = 0.f;
    for (int c = 0; c < 256; ++c) mu += xs[t][c];
    mu *= (1.f / 256.f);
    float var = 0.f;
    for (int c = 0; c < 256; ++c) { float d = xs[t][c] - mu; var += d * d; }
    var *= (1.f / 256.f);
    smu[t] = mu;
    srs[t] = rsqrtf(var + 1e-5f);
  }
  __syncthreads();
  const int r = t >> 4, c0 = t & 15;
  const float mu = smu[r], rs = srs[r];
#pragma unroll
  for (int j = 0; j < 16; ++j) {
    int c = c0 + 16 * j;
    float v = (xs[r][c] - mu) * rs * g[c] + beta[c];
    fout[(size_t)(m0 + r) * 256 + c] = v;
    if (bout) bout[(size_t)(m0 + r) * 256 + c] = (__bf16)v;
  }
}

// ---------------------------------------------------------------------------
// Host orchestration
// ---------------------------------------------------------------------------
extern "C" void kernel_launch(void* const* d_in, const int* in_sizes, int n_in,
                              void* d_out, int out_size, void* d_ws,
                              size_t ws_size, hipStream_t stream) {
  (void)in_sizes; (void)n_in; (void)out_size; (void)ws_size;
  const int N = 6000, D = 256, P = 128, NB = 16, D4 = 1024, NH = 8;

  const float* Hg  = (const float*)d_in[0];
  const int* pidx  = (const int*)d_in[1];
  const int* bas   = (const int*)d_in[2];
  const float* ipw = (const float*)d_in[4];
  const float* ipb = (const float*)d_in[5];
  const float* opw = (const float*)d_in[6];
  const float* opb = (const float*)d_in[7];
  const float* w1  = (const float*)d_in[8];
  const float* b1  = (const float*)d_in[9];
  const float* w2  = (const float*)d_in[10];
  const float* b2  = (const float*)d_in[11];
  const float* g1  = (const float*)d_in[12];
  const float* be1 = (const float*)d_in[13];
  const float* g2  = (const float*)d_in[14];
  const float* be2 = (const float*)d_in[15];
  float* out = (float*)d_out;

  char* ws = (char*)d_ws;
  size_t off = 0;
  auto alloc = [&](size_t bytes) -> char* {
    char* p = ws + off;
    off += (bytes + 255) & ~(size_t)255;
    return p;
  };
  __bf16* Hb   = (__bf16*)alloc((size_t)N * D * 2);
  __bf16* Wqkv = (__bf16*)alloc((size_t)3 * D * D * 2);
  __bf16* Wo   = (__bf16*)alloc((size_t)D * D * 2);
  __bf16* W1b  = (__bf16*)alloc((size_t)D4 * D * 2);
  __bf16* W2b  = (__bf16*)alloc((size_t)D * D4 * 2);
  __bf16* Qb   = (__bf16*)alloc((size_t)N * D * 2);
  __bf16* Hpb  = (__bf16*)alloc((size_t)P * D * 2);
  __bf16* Kb   = (__bf16*)alloc((size_t)P * D * 2);
  __bf16* Vb   = (__bf16*)alloc((size_t)P * D * 2);
  __bf16* Vt   = (__bf16*)alloc((size_t)NH * 32 * 128 * 2);
  float*  S    = (float*)alloc((size_t)NH * N * 128 * 4);
  int*    cnt  = (int*)alloc(64);
  __bf16* ctx  = (__bf16*)alloc((size_t)N * D * 2);
  float*  xf   = (float*)alloc((size_t)N * D * 4);
  __bf16* xb   = (__bf16*)alloc((size_t)N * D * 2);
  __bf16* h1   = (__bf16*)alloc((size_t)N * D4 * 2);

  const dim3 blk(256);
  auto g256 = [](int n) { return dim3((unsigned)((n + 255) / 256)); };

  // bf16 conversions of activations & weights (weights stay hot in L2)
  cvt_kernel<<<g256(N * D), blk, 0, stream>>>(Hg, Hb, N * D);
  cvt_kernel<<<g256(3 * D * D), blk, 0, stream>>>(ipw, Wqkv, 3 * D * D);
  cvt_kernel<<<g256(D * D), blk, 0, stream>>>(opw, Wo, D * D);
  cvt_kernel<<<g256(D4 * D), blk, 0, stream>>>(w1, W1b, D4 * D);
  cvt_kernel<<<g256(D * D4), blk, 0, stream>>>(w2, W2b, D * D4);
  count_kernel<<<1, 32, 0, stream>>>(bas, cnt);
  gather_kernel<<<P, D, 0, stream>>>(Hb, pidx, Hpb);

  // Projections (async-LDS pipelined WMMA GEMMs)
  gemm_async_kernel<0><<<dim3(94, 2), blk, 0, stream>>>(Hb, D, Wqkv, D, ipb,
                                                        Qb, D, N, D);
  gemm_async_kernel<0><<<dim3(2, 2), blk, 0, stream>>>(Hpb, D, Wqkv + D * D, D,
                                                       ipb + D, Kb, D, P, D);
  gemm_async_kernel<0><<<dim3(2, 2), blk, 0, stream>>>(
      Hpb, D, Wqkv + 2 * D * D, D, ipb + 2 * D, Vb, D, P, D);
  vtrans_kernel<<<g256(NH * 32 * 128), blk, 0, stream>>>(Vb, Vt);

  // Shared (batch-independent) attention scores [h,N,P]
  scores_kernel<<<dim3(N / 16, NH), blk, 0, stream>>>(Qb, Kb, S, N);

  // Per-batch: masked softmax + probs@V, out-proj+LN1, FFN1(GELU), FFN2+LN2
  for (int b = 0; b < NB; ++b) {
    attn_ctx_kernel<<<dim3(N / 16), blk, 0, stream>>>(S, Vt, bas, cnt, b, ctx,
                                                      N);
    gemm_ln_kernel<<<dim3(N / 16), blk, 0, stream>>>(ctx, D, D, Wo, opb, Hg,
                                                     g1, be1, cnt, b, xf, xb,
                                                     N);
    gemm_async_kernel<1><<<dim3(94, 8), blk, 0, stream>>>(xb, D, W1b, D, b1,
                                                          h1, D4, N, D);
    gemm_ln_kernel<<<dim3(N / 16), blk, 0, stream>>>(
        h1, D4, D4, W2b, b2, xf, g2, be2, nullptr, 0,
        out + (size_t)b * N * D, nullptr, N);
  }
}
